// Cluster_37684043055281
// MI455X (gfx1250) — matrix operand order repaired and checked by reading, not compiled, not run on hardware
//
#include <hip/hip_runtime.h>
#include <hip/hip_bf16.h>
#include <math.h>

// ---------------------------------------------------------------------------
// Sizes (match reference)
// ---------------------------------------------------------------------------
constexpr int B0    = 4;
constexpr int DIM   = 256;
constexpr int W0    = 128;
constexpr int H0    = 128;
constexpr int HW    = W0 * H0;     // 16384 pixels per (b, ch) plane
constexpr int HEADS = 8;
constexpr int HD    = 32;          // head dim (c per fold-block)
constexpr int M     = 16;          // PROP*PROP centers
constexpr int CP    = DIM / 2;     // 128 packed bf16 channel pairs
constexpr int GR    = CP / 8;      // 16 groups of 8 channel-pairs

typedef __attribute__((ext_vector_type(16))) __bf16        v16bf;
typedef __attribute__((ext_vector_type(8)))  float         v8f;
typedef __attribute__((ext_vector_type(8)))  unsigned int  v8u;

__device__ __forceinline__ float sigm(float v) {
    return 1.0f / (1.0f + __expf(-v));
}

// float -> bf16 (round to nearest even), returned as raw u16
__device__ __forceinline__ unsigned int f2bf(float f) {
    unsigned int u = __float_as_uint(f);
    unsigned int r = u + 0x7fffu + ((u >> 16) & 1u);
    return r >> 16;
}

// ---------------------------------------------------------------------------
// Intermediate layout (co-designed with the WMMA B-fragment):
//   midpk[b][g][p][s], g in [0,16), s in [0,8): dword = bf16 channel pair
//   cp = g*8+s  (channels 2cp, 2cp+1) at pixel p.
// A B-fragment for k-step kk, lane-half hi is the contiguous 32B at
//   g = kk*2+hi, s = 0..7  ->  2x global_load_b128 per fragment.
// ---------------------------------------------------------------------------

// ---------------------------------------------------------------------------
// Kernel 1: "middle" part.  value/Wv/bv are mathematically dead (split_sim==0)
// so the 7x7 conv is eliminated.  Per fold-block (bb in [0,128)):
//   cen   = 16x16 avg-pool of x-tile          (M=16, c=32)
//   sim   = sigmoid(b + a*cos(cen, x_n)) * sigmoid(b + a*cos(xyz_c, xyz_n))^2
//   dout  = (sim @ x_pts + cen) / (rowsum(sim) + 1)
//   out_n = dout^T sim                        (emitted un-folded, bf16 packed)
// plus per-batch GroupNorm sum / sumsq accumulation.
// ---------------------------------------------------------------------------
__global__ __launch_bounds__(256) void middle_kernel(
    const float* __restrict__ x, const float* __restrict__ xyz,
    const float* __restrict__ alpha_p, const float* __restrict__ beta_p,
    unsigned int* __restrict__ midpk,   // [B0][GR][HW][8] packed bf16 pairs
    float* __restrict__ gacc)           // [0..3]=sum per b, [4..7]=sumsq per b
{
    __shared__ float cenRaw[M][HD];        //  2 KB
    __shared__ float cenN[M][HD + 1];      //  2.1 KB (normalized)
    __shared__ float geoN[M][4];           //  raw then normalized
    __shared__ float simC[M][256 + 4];     // 16.6 KB sim chunk
    __shared__ float xC[256][HD + 1];      // 33.8 KB x chunk
    __shared__ float rowsumS[M];
    __shared__ float denss[M][HD];
    __shared__ float red[16];

    const int bb   = blockIdx.x;
    const int fy   = (bb >> 1) & 1;        // f1 (W-dim fold)
    const int fx   = bb & 1;               // f2 (H-dim fold)
    const int b0h  = bb >> 2;
    const int head = b0h & 7;
    const int b    = b0h >> 3;
    const int t    = threadIdx.x;

    const float alpha = alpha_p[0];
    const float beta  = beta_p[0];

    const float* xb  = x   + (size_t)(b * DIM + head * HD) * HW;
    const float* xyb = xyz + (size_t)(b * 3) * HW;

    // ---- phase 0: 16x16 pooled centers (feature + geometry) ----
    for (int pr = t; pr < M * HD; pr += 256) {
        int m = pr >> 5, c = pr & 31;
        int pw = m >> 2, ph = m & 3;
        const float* p = xb + (size_t)c * HW;
        float s = 0.f;
        for (int i = 0; i < 16; i++) {
            int off = (fy * 64 + pw * 16 + i) * W0 + fx * 64 + ph * 16;
            #pragma unroll
            for (int j = 0; j < 16; j++) s += p[off + j];
        }
        cenRaw[m][c] = s * (1.f / 256.f);
    }
    if (t < M * 3) {
        int m = t / 3, ch = t % 3;
        int pw = m >> 2, ph = m & 3;
        const float* p = xyb + (size_t)ch * HW;
        float s = 0.f;
        for (int i = 0; i < 16; i++) {
            int off = (fy * 64 + pw * 16 + i) * W0 + fx * 64 + ph * 16;
            #pragma unroll
            for (int j = 0; j < 16; j++) s += p[off + j];
        }
        geoN[m][ch] = s * (1.f / 256.f);
    }
    __syncthreads();
    if (t < M) {
        float nn = 0.f;
        #pragma unroll
        for (int c = 0; c < HD; c++) nn += cenRaw[t][c] * cenRaw[t][c];
        float inv = 1.f / fmaxf(sqrtf(nn), 1e-12f);
        #pragma unroll
        for (int c = 0; c < HD; c++) cenN[t][c] = cenRaw[t][c] * inv;
        float g = geoN[t][0]*geoN[t][0] + geoN[t][1]*geoN[t][1] + geoN[t][2]*geoN[t][2];
        float gi = 1.f / fmaxf(sqrtf(g), 1e-12f);
        geoN[t][0] *= gi; geoN[t][1] *= gi; geoN[t][2] *= gi;
    }
    __syncthreads();

    // ---- phase 1: sim + rowsum + acc (= sim @ x_pts), chunked over n ----
    const int cL = t & 31;        // this thread's channel for acc
    const int mL = t >> 5;        // this thread's m (and m+8) for acc
    float accA = 0.f, accB = 0.f; // acc[mL][cL], acc[mL+8][cL]
    float rs = 0.f;               // rowsum partial (threads 0..15)

    for (int ch = 0; ch < 16; ch++) {
        const int n  = ch * 256 + t;
        const int wi = n >> 6, hi = n & 63;
        const size_t poff = (size_t)(fy * 64 + wi) * W0 + fx * 64 + hi;

        float xn[HD]; float nn = 0.f;
        #pragma unroll
        for (int c = 0; c < HD; c++) {
            float v = xb[(size_t)c * HW + poff];
            xn[c] = v; xC[t][c] = v; nn += v * v;
        }
        float inv = 1.f / fmaxf(sqrtf(nn), 1e-12f);
        float p0 = xyb[poff], p1 = xyb[HW + poff], p2 = xyb[2 * HW + poff];
        float gi = 1.f / fmaxf(sqrtf(p0*p0 + p1*p1 + p2*p2), 1e-12f);

        #pragma unroll
        for (int m = 0; m < M; m++) {
            float d = 0.f;
            #pragma unroll
            for (int c = 0; c < HD; c++) d += cenN[m][c] * xn[c];
            float s  = sigm(beta + alpha * d * inv);
            float gd = geoN[m][0]*p0 + geoN[m][1]*p1 + geoN[m][2]*p2;
            float g  = sigm(beta + alpha * gd * gi);
            simC[m][t] = s * g * g;
        }
        __syncthreads();

        for (int n2 = 0; n2 < 256; n2++) {
            float xv = xC[n2][cL];
            accA = fmaf(simC[mL][n2],     xv, accA);
            accB = fmaf(simC[mL + 8][n2], xv, accB);
        }
        if (t < M) {
            float r = 0.f;
            for (int n2 = 0; n2 < 256; n2++) r += simC[t][n2];
            rs += r;
        }
        __syncthreads();
    }
    if (t < M) rowsumS[t] = rs;
    __syncthreads();
    denss[mL][cL]     = (accA + cenRaw[mL][cL])     / (rowsumS[mL] + 1.f);
    denss[mL + 8][cL] = (accB + cenRaw[mL + 8][cL]) / (rowsumS[mL + 8] + 1.f);
    __syncthreads();

    // ---- phase 2: recompute sim, emit out = denss^T sim, bf16 packed ----
    float gs = 0.f, gss = 0.f;
    for (int ch = 0; ch < 16; ch++) {
        const int n  = ch * 256 + t;
        const int wi = n >> 6, hi = n & 63;
        const size_t poff = (size_t)(fy * 64 + wi) * W0 + fx * 64 + hi;

        float xn[HD]; float nn = 0.f;
        #pragma unroll
        for (int c = 0; c < HD; c++) {
            float v = xb[(size_t)c * HW + poff];
            xn[c] = v; nn += v * v;
        }
        float inv = 1.f / fmaxf(sqrtf(nn), 1e-12f);
        float p0 = xyb[poff], p1 = xyb[HW + poff], p2 = xyb[2 * HW + poff];
        float gi = 1.f / fmaxf(sqrtf(p0*p0 + p1*p1 + p2*p2), 1e-12f);

        float o[HD];
        #pragma unroll
        for (int c = 0; c < HD; c++) o[c] = 0.f;
        #pragma unroll
        for (int m = 0; m < M; m++) {
            float d = 0.f;
            #pragma unroll
            for (int c = 0; c < HD; c++) d += cenN[m][c] * xn[c];
            float s  = sigm(beta + alpha * d * inv);
            float gd = geoN[m][0]*p0 + geoN[m][1]*p1 + geoN[m][2]*p2;
            float g  = sigm(beta + alpha * gd * gi);
            float sm = s * g * g;
            #pragma unroll
            for (int c = 0; c < HD; c++) o[c] = fmaf(denss[m][c], sm, o[c]);
        }
        // pack 32 channels -> 16 bf16-pair dwords -> two 32B grouped stores
        v8u pk0, pk1;
        #pragma unroll
        for (int j = 0; j < 8; j++) {
            float a0 = o[2 * j],      a1 = o[2 * j + 1];
            float b0 = o[16 + 2 * j], b1 = o[17 + 2 * j];
            gs  += a0 + a1 + b0 + b1;
            gss += a0 * a0 + a1 * a1 + b0 * b0 + b1 * b1;
            pk0[j] = f2bf(a0) | (f2bf(a1) << 16);
            pk1[j] = f2bf(b0) | (f2bf(b1) << 16);
        }
        unsigned int* dst0 = midpk + ((size_t)(b * GR + head * 2)     * HW + poff) * 8;
        unsigned int* dst1 = midpk + ((size_t)(b * GR + head * 2 + 1) * HW + poff) * 8;
        *(v8u*)dst0 = pk0;
        *(v8u*)dst1 = pk1;
    }

    // ---- GroupNorm partials: wave reduce then LDS reduce then global atomics
    for (int off = 16; off > 0; off >>= 1) {
        gs  += __shfl_down(gs, off);
        gss += __shfl_down(gss, off);
    }
    if ((t & 31) == 0) { red[mL] = gs; red[8 + mL] = gss; }
    __syncthreads();
    if (t == 0) {
        float a = 0.f, q = 0.f;
        #pragma unroll
        for (int w = 0; w < 8; w++) { a += red[w]; q += red[8 + w]; }
        atomicAdd(&gacc[b], a);
        atomicAdd(&gacc[4 + b], q);
    }
}

// ---------------------------------------------------------------------------
// Kernel 2: fold GroupNorm affine into projection weights.
//   W'[b][o][c] = Wproj[o][c] * (inv_b * gn_w[c])         (bf16 pair packed)
//   bias'[b][o] = sum_c Wproj[o][c]*(gn_b[c]-mu_b*inv_b*gn_w[c]) + bproj[o]
// ---------------------------------------------------------------------------
__global__ __launch_bounds__(256) void prep_proj(
    const float* __restrict__ Wproj, const float* __restrict__ bproj,
    const float* __restrict__ gnw,   const float* __restrict__ gnb,
    const float* __restrict__ gacc,
    unsigned int* __restrict__ Wpk,  // [B0][DIM][CP]
    float* __restrict__ biasp)       // [B0][DIM]
{
    const int b = blockIdx.x;
    const int o = threadIdx.x;
    const float cnt = (float)DIM * (float)HW;
    const float mu  = gacc[b] / cnt;
    const float var = gacc[4 + b] / cnt - mu * mu;
    const float inv = rsqrtf(var + 1e-5f);

    const float* wr = Wproj + (size_t)o * DIM;
    float bs = bproj[o];
    for (int cc = 0; cc < CP; cc++) {
        int c0 = 2 * cc, c1 = 2 * cc + 1;
        float a0 = inv * gnw[c0], a1 = inv * gnw[c1];
        float d0 = gnb[c0] - mu * a0, d1 = gnb[c1] - mu * a1;
        float w0 = wr[c0], w1 = wr[c1];
        bs += w0 * d0 + w1 * d1;
        Wpk[(size_t)(b * DIM + o) * CP + cc] = f2bf(w0 * a0) | (f2bf(w1 * a1) << 16);
    }
    biasp[b * DIM + o] = bs;
}

// ---------------------------------------------------------------------------
// Kernel 3: 1x1 projection as bf16 WMMA GEMM + fused SiLU.
//   out[b][o][p] = silu( sum_c W'[b][o][c] * mid[b][c][p] + bias'[b][o] )
//
// One wave owns a 32(M) x 32(N) macro-tile = 2x2 grid of 16x16 WMMA tiles.
// Fully unrolled K=256 (8 steps):  per step 4x b128 (A) + 4x b128 (B)
// feed 4 independent v_wmma_f32_16x16x32_bf16 -> 32 static WMMAs/wave.
// Fragments use contiguous k = 16*hi + s for both A and B (16-bit A 16x32, B 32x16);
// the B operand is stored pre-swizzled so a fragment is one 32B run.
// ---------------------------------------------------------------------------
__global__ __launch_bounds__(256) void proj_wmma(
    const unsigned int* __restrict__ Wpk,
    const float* __restrict__ biasp,
    const unsigned int* __restrict__ midpk,
    float* __restrict__ out)
{
    const int wid  = threadIdx.x >> 5;
    const int lane = threadIdx.x & 31;
    const int tile = blockIdx.x * 8 + wid;      // 16384 macro-tiles total
    const int nt2  = tile & 511;                // 512 N-macrotiles (32 px each)
    const int mt2  = (tile >> 9) & 7;           // 8 M-macrotiles (32 ch each)
    const int b    = tile >> 12;
    const int nb   = nt2 * 32;
    const int row  = lane & 15;
    const int hi   = lane >> 4;                 // 0: lanes 0-15, 1: lanes 16-31

    // A rows for the two M-subtiles (bf16 pairs, row-major over K)
    const unsigned int* arow0 = Wpk + (size_t)(b * DIM + mt2 * 32 + row) * CP;
    const unsigned int* arow1 = arow0 + (size_t)16 * CP;
    // B: grouped layout [b][g][p][8]; fragment g = kk*2 + hi at pixel nb+row
    const unsigned int* bbase = midpk + ((size_t)(b * GR + hi) * HW + nb + row) * 8;

    v8f acc00 = {}, acc01 = {}, acc10 = {}, acc11 = {};

    #pragma unroll
    for (int kk = 0; kk < 8; kk++) {
        v8u au0, au1;
        // A: lane holds M=row; VGPR v -> K-pair v + hi*8
        #pragma unroll
        for (int v = 0; v < 8; v++) {
            int pair = v + hi * 8;
            au0[v] = arow0[kk * 16 + pair];
            au1[v] = arow1[kk * 16 + pair];
        }
        // B: one contiguous 32B run per fragment (pairs v=0..7 of group kk*2+hi)
        const unsigned int* bp = bbase + (size_t)(kk * 2) * HW * 8;
        v8u bu0 = *(const v8u*)bp;           // N-subtile 0 (pixel nb+row)
        v8u bu1 = *(const v8u*)(bp + 128);   // N-subtile 1 (pixel nb+16+row)

        v16bf a0 = __builtin_bit_cast(v16bf, au0);
        v16bf a1 = __builtin_bit_cast(v16bf, au1);
        v16bf b0 = __builtin_bit_cast(v16bf, bu0);
        v16bf b1 = __builtin_bit_cast(v16bf, bu1);
        acc00 = __builtin_amdgcn_wmma_f32_16x16x32_bf16(false, a0, false, b0, (short)0, acc00, false, false);
        acc01 = __builtin_amdgcn_wmma_f32_16x16x32_bf16(false, a0, false, b1, (short)0, acc01, false, false);
        acc10 = __builtin_amdgcn_wmma_f32_16x16x32_bf16(false, a1, false, b0, (short)0, acc10, false, false);
        acc11 = __builtin_amdgcn_wmma_f32_16x16x32_bf16(false, a1, false, b1, (short)0, acc11, false, false);
    }

    // C/D layout: VGPR r, lanes 0-15 -> (M=r, N=lane); lanes 16-31 -> (M=r+8)
    #pragma unroll
    for (int ms = 0; ms < 2; ms++) {
        #pragma unroll
        for (int r = 0; r < 8; r++) {
            int och = mt2 * 32 + ms * 16 + r + hi * 8;
            float bia = biasp[b * DIM + och];
            float v0 = (ms ? acc10[r] : acc00[r]) + bia;
            float v1 = (ms ? acc11[r] : acc01[r]) + bia;
            v0 = v0 * (1.0f / (1.0f + __expf(-v0)));    // SiLU
            v1 = v1 * (1.0f / (1.0f + __expf(-v1)));
            float* orow = out + (size_t)(b * DIM + och) * HW + nb + row;
            orow[0]  = v0;
            orow[16] = v1;
        }
    }
}

// ---------------------------------------------------------------------------
// Host launcher
// ---------------------------------------------------------------------------
extern "C" void kernel_launch(void* const* d_in, const int* in_sizes, int n_in,
                              void* d_out, int out_size, void* d_ws, size_t ws_size,
                              hipStream_t stream) {
    const float* x     = (const float*)d_in[0];
    const float* xyz   = (const float*)d_in[1];
    // d_in[2] = Wv, d_in[3] = bv : provably dead (split_sim == 0 in reference)
    const float* Wproj = (const float*)d_in[4];
    const float* bproj = (const float*)d_in[5];
    const float* gnw   = (const float*)d_in[6];
    const float* gnb   = (const float*)d_in[7];
    const float* alpha = (const float*)d_in[8];
    const float* beta  = (const float*)d_in[9];
    float* out = (float*)d_out;

    char* ws = (char*)d_ws;
    float*        gacc  = (float*)ws;                          //   32 B (pad 256)
    unsigned int* Wpk   = (unsigned int*)(ws + 256);           //  512 KB
    float*        biasp = (float*)(ws + 256 + (size_t)B0 * DIM * CP * 4);
    unsigned int* midpk = (unsigned int*)(ws + 256 + (size_t)B0 * DIM * CP * 4
                                             + (size_t)B0 * DIM * 4); // 33.5 MB

    hipMemsetAsync(gacc, 0, 8 * sizeof(float), stream);
    middle_kernel<<<128, 256, 0, stream>>>(x, xyz, alpha, beta, midpk, gacc);
    prep_proj<<<B0, 256, 0, stream>>>(Wproj, bproj, gnw, gnb, gacc, Wpk, biasp);
    proj_wmma<<<2048, 256, 0, stream>>>(Wpk, biasp, midpk, out);
}
